// RFFGPA_LAYER_39204461478735
// MI455X (gfx1250) — matrix operand order repaired and checked by reading, not compiled, not run on hardware
//
#include <hip/hip_runtime.h>
#include <hip/hip_bf16.h>
#include <math.h>

// ---------------------------------------------------------------------------
// RFF-GPA layer for MI455X (gfx1250, wave32, WMMA + TDM async tensor loads).
// Heavy GEMMs: v_wmma_f32_16x16x32_bf16, weight tiles staged to LDS by the
// Tensor Data Mover (tensor_load_to_lds / s_wait_tensorcnt), double buffered.
// Cholesky + triangular solves stay scalar f32 in 256KB LDS (320KB/WGP).
// ---------------------------------------------------------------------------

typedef __attribute__((ext_vector_type(16))) __bf16 bf16x16;
typedef __attribute__((ext_vector_type(8)))  __bf16 bf16x8;
typedef __attribute__((ext_vector_type(8)))  float  f32x8;
typedef unsigned int u32x4 __attribute__((ext_vector_type(4)));
typedef int          i32x8 __attribute__((ext_vector_type(8)));
typedef int          i32x4 __attribute__((ext_vector_type(4)));

#define H_    16
#define VD_   64
#define M_    128
#define M2_   256
#define B_    4
#define L_    2048
#define HD_   1024
#define BL_   8192
#define KC_   128          // K-chunk staged per TDM transfer
#define RIDGE_ 0.01f
#define SCALE_ 0.08838834764831845f   // 1/sqrt(M)

// ---- WMMA fragment loaders (wave32 layouts, CDNA5 ISA 7.12.2) ---------------
static __device__ __forceinline__ bf16x16
load_a_frag(const __bf16* A, int lda, int m0, int k0, int lane) {
  const int m    = m0 + (lane & 15);
  const int koff = (lane < 16) ? 0 : 8;
  const __bf16* p = A + (long)m * lda + k0 + koff;
  union { bf16x16 v; bf16x8 h[2]; } u;
  u.h[0] = *(const bf16x8*)(p);
  u.h[1] = *(const bf16x8*)(p + 16);
  return u.v;
}

// B[k][n] = Wt[n][k]; lane n holds k = 8h+[0..7] and 16+8h+[0..7].
static __device__ __forceinline__ bf16x16
load_b_frag(const __bf16* Wt, int ldw, int n0, int k0, int lane) {
  const int n  = n0 + (lane & 15);
  const int kk = k0 + ((lane < 16) ? 0 : 8);
  const __bf16* p = Wt + (long)n * ldw + kk;
  union { bf16x16 v; bf16x8 h[2]; } u;
  u.h[0] = *(const bf16x8*)(p);
  u.h[1] = *(const bf16x8*)(p + 16);
  return u.v;
}

static __device__ __forceinline__ f32x8 wmma_bf16(bf16x16 a, bf16x16 b, f32x8 c) {
  return __builtin_amdgcn_wmma_f32_16x16x32_bf16(false, a, false, b,
                                                 (short)0, c, false, false);
}

// ---- TDM staging: 64 rows x 128 cols bf16 tile, global(row stride HD_)->LDS -
#if __has_builtin(__builtin_amdgcn_tensor_load_to_lds)
#define USE_TDM_ 1
static __device__ __forceinline__ void
stage_w_tile(const __bf16* gsrc, __bf16* sdst, int lane) {
  (void)lane;
  const unsigned long long ga = (unsigned long long)(const void*)gsrc;
  const unsigned lds = (unsigned)(unsigned long long)(const void*)sdst;
  u32x4 g0;
  g0[0] = 1u;                                  // count=1 user descriptor
  g0[1] = lds;                                 // lds_addr [63:32]
  g0[2] = (unsigned)(ga & 0xffffffffu);        // global_addr lo
  g0[3] = (unsigned)((ga >> 32) & 0x01ffffffu) | (2u << 30);  // addr hi | type=2
  i32x8 g1;
  g1[0] = (int)(1u << 16);                     // data_size=1 -> 2 bytes/elem
  g1[1] = (int)((unsigned)HD_ << 16);          // tensor_dim0[15:0] @ [63:48]
  g1[2] = (int)(((unsigned)HD_ >> 16) | ((unsigned)HD_ << 16)); // d0 hi | tensor_dim1 lo
  g1[3] = (int)(((unsigned)HD_ >> 16) | ((unsigned)KC_ << 16)); // d1 hi | tile_dim0
  g1[4] = (int)(64u);                          // tile_dim1 = 64 rows
  g1[5] = (int)((unsigned)HD_);                // tensor_dim0_stride lo32
  g1[6] = 0;
  g1[7] = 0;
  i32x4 z4 = {0, 0, 0, 0};
#if __clang_major__ >= 23
  i32x8 z8 = {0, 0, 0, 0, 0, 0, 0, 0};
  __builtin_amdgcn_tensor_load_to_lds(g0, g1, z4, z4, z8, 0);
#else
  __builtin_amdgcn_tensor_load_to_lds(g0, g1, z4, z4, 0);
#endif
}
static __device__ __forceinline__ void wait_stage(int pending) {
#if __has_builtin(__builtin_amdgcn_s_wait_tensorcnt)
  if (pending) __builtin_amdgcn_s_wait_tensorcnt(1);
  else         __builtin_amdgcn_s_wait_tensorcnt(0);
#endif
}
#else
#define USE_TDM_ 0
static __device__ __forceinline__ void
stage_w_tile(const __bf16* gsrc, __bf16* sdst, int lane) {
  for (int i = lane; i < 64 * KC_ / 8; i += 32) {
    const int row = i >> 4, c = (i & 15) * 8;
    *(bf16x8*)(sdst + row * KC_ + c) = *(const bf16x8*)(gsrc + (long)row * HD_ + c);
  }
}
static __device__ __forceinline__ void wait_stage(int pending) { (void)pending; }
#endif

// ---- converts ---------------------------------------------------------------
__global__ void k_cvt_bf16(const float* __restrict__ src, __bf16* __restrict__ dst, int n) {
  int i = blockIdx.x * 256 + threadIdx.x;
  if (i < n) dst[i] = (__bf16)src[i];
}

// rff_W (H,VD,M) -> bf16 transposed (H,M,VD)
__global__ void k_cvt_rffW(const float* __restrict__ src, __bf16* __restrict__ dst) {
  int i = blockIdx.x * 256 + threadIdx.x;
  if (i < H_ * VD_ * M_) {
    int h = i / (VD_ * M_);
    int r = i % (VD_ * M_);
    int d = r / M_;
    int m = r % M_;
    dst[((long)h * M_ + m) * VD_ + d] = (__bf16)src[i];
  }
}

// ---- big projection GEMM: C(8192x1024) = A @ W^T, TDM-staged weights --------
// mode 0: store per-head row-major [bh][l][d]  bf16   (q, k)
// mode 2: store transposed+masked  [bh][d][l]  bf16   (v)
// mode 3: f32 out = (acc + bo[n]) * mask[row]         (final Wo projection)
__global__ void __launch_bounds__(128)
k_proj(const __bf16* __restrict__ X, const __bf16* __restrict__ W,
       const float* __restrict__ mask, const float* __restrict__ bo,
       __bf16* __restrict__ outb, float* __restrict__ outf, int mode) {
  __shared__ __bf16 sW[2][64 * KC_];          // 2 x 16KB double buffer
  const int lane = threadIdx.x & 31;
  const int wave = threadIdx.x >> 5;
  const int m0 = blockIdx.y * 128 + wave * 32;  // 32 rows per wave
  const int n0 = blockIdx.x * 64;               // 64 cols per block
  const __bf16* Wb = W + (long)n0 * HD_;
  f32x8 acc[2][4] = {};

  const int NCH = HD_ / KC_;
  if (wave == 0) stage_w_tile(Wb, sW[0], lane);
  for (int kc = 0; kc < NCH; ++kc) {
    if (wave == 0) {
      if (kc + 1 < NCH) {
        stage_w_tile(Wb + (kc + 1) * KC_, sW[(kc + 1) & 1], lane);
        wait_stage(1);
      } else {
        wait_stage(0);
      }
    }
    __syncthreads();
    const __bf16* tile = sW[kc & 1];
    const int kbase = kc * KC_;
    if (kc + 1 < NCH)
      __builtin_prefetch(X + (long)(m0 + lane) * HD_ + kbase + KC_, 0, 0);
#pragma unroll
    for (int ks = 0; ks < KC_ / 32; ++ks) {
      const int k0 = kbase + ks * 32;
      bf16x16 a0 = load_a_frag(X, HD_, m0,      k0, lane);
      bf16x16 a1 = load_a_frag(X, HD_, m0 + 16, k0, lane);
#pragma unroll
      for (int t = 0; t < 4; ++t) {
        bf16x16 b = load_b_frag(tile, KC_, 16 * t, ks * 32, lane);
        acc[0][t] = wmma_bf16(a0, b, acc[0][t]);
        acc[1][t] = wmma_bf16(a1, b, acc[1][t]);
      }
    }
    __syncthreads();
  }

#pragma unroll
  for (int rt = 0; rt < 2; ++rt) {
    const int rowbase = m0 + rt * 16 + ((lane < 16) ? 0 : 8);
    const int b = rowbase >> 11, l = rowbase & (L_ - 1);
#pragma unroll
    for (int t = 0; t < 4; ++t) {
      const int n = n0 + 16 * t + (lane & 15);
      if (mode == 3) {
        const float bn = bo[n];
#pragma unroll
        for (int r = 0; r < 8; ++r)
          outf[(long)(rowbase + r) * HD_ + n] = (acc[rt][t][r] + bn) * mask[rowbase + r];
      } else if (mode == 2) {
        const int h = n >> 6, d = n & 63;
        union { bf16x8 v; __bf16 e[8]; } tmp;
#pragma unroll
        for (int r = 0; r < 8; ++r) tmp.e[r] = (__bf16)(acc[rt][t][r] * mask[rowbase + r]);
        *(bf16x8*)(outb + ((long)((b * H_ + h) * VD_ + d)) * L_ + l) = tmp.v;
      } else {
        const int h = n >> 6, d = n & 63;
#pragma unroll
        for (int r = 0; r < 8; ++r)
          outb[((long)((b * H_ + h) * L_ + l + r)) * VD_ + d] = (__bf16)acc[rt][t][r];
      }
    }
  }
}

// ---- RFF: proj = Xh @ rffWt^T + b;  Phi = scale*[cos, sin] ------------------
__global__ void k_rff(const __bf16* __restrict__ Xh, const __bf16* __restrict__ rffWt,
                      const float* __restrict__ rffb, const float* __restrict__ mask,
                      __bf16* __restrict__ out, int transpose_mask) {
  const int lane = threadIdx.x & 31;
  const int wave = threadIdx.x >> 5;
  const int bh = blockIdx.z;
  const int h = bh & (H_ - 1);
  const int m0 = blockIdx.y * 64 + wave * 16;
  const int n0 = blockIdx.x * 64;
  const __bf16* A  = Xh + (long)bh * L_ * VD_;
  const __bf16* Bw = rffWt + (long)h * M_ * VD_;
  f32x8 acc[4] = {};
  for (int k0 = 0; k0 < VD_; k0 += 32) {
    bf16x16 a = load_a_frag(A, VD_, m0, k0, lane);
#pragma unroll
    for (int t = 0; t < 4; ++t) {
      bf16x16 b = load_b_frag(Bw, VD_, n0 + 16 * t, k0, lane);
      acc[t] = wmma_bf16(a, b, acc[t]);
    }
  }
  const int rowbase = m0 + ((lane < 16) ? 0 : 8);
  const int mflat = (bh >> 4) * L_ + rowbase;
#pragma unroll
  for (int t = 0; t < 4; ++t) {
    const int n = n0 + 16 * t + (lane & 15);
    const float bias = rffb[h * M_ + n];
    if (transpose_mask) {
      union { bf16x8 v; __bf16 e[8]; } tc, ts;
#pragma unroll
      for (int r = 0; r < 8; ++r) {
        float p = acc[t][r] + bias;
        float mk = mask[mflat + r] * SCALE_;
        tc.e[r] = (__bf16)(__cosf(p) * mk);
        ts.e[r] = (__bf16)(__sinf(p) * mk);
      }
      *(bf16x8*)(out + ((long)(bh * M2_ + n)) * L_ + rowbase) = tc.v;
      *(bf16x8*)(out + ((long)(bh * M2_ + M_ + n)) * L_ + rowbase) = ts.v;
    } else {
#pragma unroll
      for (int r = 0; r < 8; ++r) {
        float p = acc[t][r] + bias;
        long base = ((long)bh * L_ + rowbase + r) * M2_;
        out[base + n]      = (__bf16)(__cosf(p) * SCALE_);
        out[base + M_ + n] = (__bf16)(__sinf(p) * SCALE_);
      }
    }
  }
}

// ---- Gram: G = PhiK^T PhiK + ridge*I (256x256 per bh, K = 2048) -------------
__global__ void __launch_bounds__(128)
k_gram(const __bf16* __restrict__ PhiKt, float* __restrict__ G) {
  const int lane = threadIdx.x & 31;
  const int wave = threadIdx.x >> 5;
  const int bh = blockIdx.z;
  const int m0 = blockIdx.y * 128 + wave * 32;
  const int n0 = blockIdx.x * 64;
  const __bf16* P = PhiKt + (long)bh * M2_ * L_;
  f32x8 acc[2][4] = {};
  for (int k0 = 0; k0 < L_; k0 += 32) {
    bf16x16 a0 = load_a_frag(P, L_, m0,      k0, lane);
    bf16x16 a1 = load_a_frag(P, L_, m0 + 16, k0, lane);
#pragma unroll
    for (int t = 0; t < 4; ++t) {
      bf16x16 b = load_b_frag(P, L_, n0 + 16 * t, k0, lane);
      acc[0][t] = wmma_bf16(a0, b, acc[0][t]);
      acc[1][t] = wmma_bf16(a1, b, acc[1][t]);
    }
  }
#pragma unroll
  for (int rt = 0; rt < 2; ++rt) {
    const int rowbase = m0 + rt * 16 + ((lane < 16) ? 0 : 8);
#pragma unroll
    for (int t = 0; t < 4; ++t) {
      const int n = n0 + 16 * t + (lane & 15);
#pragma unroll
      for (int r = 0; r < 8; ++r) {
        const int row = rowbase + r;
        G[((long)bh * M2_ + row) * M2_ + n] = acc[rt][t][r] + ((row == n) ? RIDGE_ : 0.0f);
      }
    }
  }
}

// ---- T = PhiK^T V (256x64 per bh, K = 2048) ---------------------------------
__global__ void k_tmat(const __bf16* __restrict__ PhiKt, const __bf16* __restrict__ Vt,
                       float* __restrict__ T) {
  const int lane = threadIdx.x & 31;
  const int wave = threadIdx.x >> 5;
  const int bh = blockIdx.z;
  const int m0 = blockIdx.y * 64 + wave * 16;
  const __bf16* P = PhiKt + (long)bh * M2_ * L_;
  const __bf16* V = Vt + (long)bh * VD_ * L_;
  f32x8 acc[4] = {};
  for (int k0 = 0; k0 < L_; k0 += 32) {
    bf16x16 a = load_a_frag(P, L_, m0, k0, lane);
#pragma unroll
    for (int t = 0; t < 4; ++t) {
      bf16x16 b = load_b_frag(V, L_, 16 * t, k0, lane);
      acc[t] = wmma_bf16(a, b, acc[t]);
    }
  }
  const int rowbase = m0 + ((lane < 16) ? 0 : 8);
#pragma unroll
  for (int t = 0; t < 4; ++t) {
    const int n = 16 * t + (lane & 15);
#pragma unroll
    for (int r = 0; r < 8; ++r)
      T[((long)bh * M2_ + rowbase + r) * VD_ + n] = acc[t][r];
  }
}

// ---- Cholesky(256) + two triangular solves per bh ---------------------------
__global__ void k_chol(float* __restrict__ G, float* __restrict__ T,
                       __bf16* __restrict__ Wsolt) {
  extern __shared__ float Ls[];
  const int bh = blockIdx.x;
  const int tid = threadIdx.x;  // 256
  float* Gb = G + (long)bh * M2_ * M2_;
  float* Tb = T + (long)bh * M2_ * VD_;
  for (int i = tid; i < M2_ * M2_; i += 256) Ls[i] = Gb[i];
  __syncthreads();
  for (int k = 0; k < M2_; ++k) {
    if (tid == 0) Ls[k * M2_ + k] = sqrtf(Ls[k * M2_ + k]);
    __syncthreads();
    const float dk = Ls[k * M2_ + k];
    for (int i = k + 1 + tid; i < M2_; i += 256) Ls[i * M2_ + k] /= dk;
    __syncthreads();
    for (int j = k + 1; j < M2_; ++j) {
      const float ljk = Ls[j * M2_ + k];
      for (int i = j + tid; i < M2_; i += 256)
        Ls[i * M2_ + j] -= Ls[i * M2_ + k] * ljk;
    }
    __syncthreads();
  }
  if (tid < VD_) {
    const int d = tid;
    for (int m = 0; m < M2_; ++m) {
      float s = Tb[m * VD_ + d];
      for (int j = 0; j < m; ++j) s -= Ls[m * M2_ + j] * Tb[j * VD_ + d];
      Tb[m * VD_ + d] = s / Ls[m * M2_ + m];
    }
    for (int m = M2_ - 1; m >= 0; --m) {
      float s = Tb[m * VD_ + d];
      for (int j = m + 1; j < M2_; ++j) s -= Ls[j * M2_ + m] * Tb[j * VD_ + d];
      const float w = s / Ls[m * M2_ + m];
      Tb[m * VD_ + d] = w;
      Wsolt[((long)bh * VD_ + d) * M2_ + m] = (__bf16)w;
    }
  }
}

// ---- out = PhiQ @ W (2048x64 per bh, K = 256) -> (B,L,HD) bf16 --------------
__global__ void k_outp(const __bf16* __restrict__ PhiQ, const __bf16* __restrict__ Wsolt,
                       __bf16* __restrict__ Y) {
  const int lane = threadIdx.x & 31;
  const int wave = threadIdx.x >> 5;
  const int bh = blockIdx.z;
  const int m0 = blockIdx.y * 64 + wave * 16;
  const __bf16* A  = PhiQ + (long)bh * L_ * M2_;
  const __bf16* Bw = Wsolt + (long)bh * VD_ * M2_;
  f32x8 acc[4] = {};
  for (int k0 = 0; k0 < M2_; k0 += 32) {
    bf16x16 a = load_a_frag(A, M2_, m0, k0, lane);
#pragma unroll
    for (int t = 0; t < 4; ++t) {
      bf16x16 b = load_b_frag(Bw, M2_, 16 * t, k0, lane);
      acc[t] = wmma_bf16(a, b, acc[t]);
    }
  }
  const int rowbase = m0 + ((lane < 16) ? 0 : 8);
  const int b = bh >> 4, h = bh & (H_ - 1);
#pragma unroll
  for (int t = 0; t < 4; ++t) {
    const int n = 16 * t + (lane & 15);
#pragma unroll
    for (int r = 0; r < 8; ++r)
      Y[((long)(b * L_ + rowbase + r)) * HD_ + h * VD_ + n] = (__bf16)acc[t][r];
  }
}

// ---------------------------------------------------------------------------
extern "C" void kernel_launch(void* const* d_in, const int* in_sizes, int n_in,
                              void* d_out, int out_size, void* d_ws, size_t ws_size,
                              hipStream_t stream) {
  (void)in_sizes; (void)n_in; (void)out_size; (void)ws_size;
  const float* x    = (const float*)d_in[0];
  const float* mask = (const float*)d_in[2];
  const float* Wq   = (const float*)d_in[3];
  const float* Wk   = (const float*)d_in[4];
  const float* Wv   = (const float*)d_in[5];
  const float* Wo   = (const float*)d_in[6];
  const float* bo   = (const float*)d_in[7];
  const float* rffW = (const float*)d_in[8];
  const float* rffb = (const float*)d_in[9];
  float* out = (float*)d_out;

  char* w = (char*)d_ws;
  size_t off = 0;
  auto alloc = [&](size_t bytes) {
    void* p = w + off;
    off += (bytes + 255) & ~(size_t)255;
    return p;
  };
  __bf16* xb     = (__bf16*)alloc((size_t)BL_ * HD_ * 2);
  __bf16* Wqb    = (__bf16*)alloc((size_t)HD_ * HD_ * 2);
  __bf16* Wkb    = (__bf16*)alloc((size_t)HD_ * HD_ * 2);
  __bf16* Wvb    = (__bf16*)alloc((size_t)HD_ * HD_ * 2);
  __bf16* Wob    = (__bf16*)alloc((size_t)HD_ * HD_ * 2);
  __bf16* rffWtb = (__bf16*)alloc((size_t)H_ * M_ * VD_ * 2);
  __bf16* qh     = (__bf16*)alloc((size_t)B_ * H_ * L_ * VD_ * 2);
  __bf16* kh     = (__bf16*)alloc((size_t)B_ * H_ * L_ * VD_ * 2);
  __bf16* vt     = (__bf16*)alloc((size_t)B_ * H_ * VD_ * L_ * 2);
  __bf16* phiQ   = (__bf16*)alloc((size_t)B_ * H_ * L_ * M2_ * 2);
  __bf16* phiKt  = (__bf16*)alloc((size_t)B_ * H_ * M2_ * L_ * 2);
  float*  G      = (float*) alloc((size_t)B_ * H_ * M2_ * M2_ * 4);
  float*  Tm     = (float*) alloc((size_t)B_ * H_ * M2_ * VD_ * 4);
  __bf16* Wsolt  = (__bf16*)alloc((size_t)B_ * H_ * VD_ * M2_ * 2);
  __bf16* yb     = (__bf16*)alloc((size_t)BL_ * HD_ * 2);

  int n;
  n = BL_ * HD_; k_cvt_bf16<<<(n + 255) / 256, 256, 0, stream>>>(x, xb, n);
  n = HD_ * HD_;
  k_cvt_bf16<<<(n + 255) / 256, 256, 0, stream>>>(Wq, Wqb, n);
  k_cvt_bf16<<<(n + 255) / 256, 256, 0, stream>>>(Wk, Wkb, n);
  k_cvt_bf16<<<(n + 255) / 256, 256, 0, stream>>>(Wv, Wvb, n);
  k_cvt_bf16<<<(n + 255) / 256, 256, 0, stream>>>(Wo, Wob, n);
  n = H_ * VD_ * M_;
  k_cvt_rffW<<<(n + 255) / 256, 256, 0, stream>>>(rffW, rffWtb);

  dim3 blk(128);  // 4 waves
  dim3 g_proj(HD_ / 64, BL_ / 128);
  k_proj<<<g_proj, blk, 0, stream>>>(xb, Wqb, mask, bo, qh, nullptr, 0);
  k_proj<<<g_proj, blk, 0, stream>>>(xb, Wkb, mask, bo, kh, nullptr, 0);
  k_proj<<<g_proj, blk, 0, stream>>>(xb, Wvb, mask, bo, vt, nullptr, 2);

  dim3 g_rff(M_ / 64, L_ / 64, B_ * H_);
  k_rff<<<g_rff, blk, 0, stream>>>(qh, rffWtb, rffb, mask, phiQ, 0);
  k_rff<<<g_rff, blk, 0, stream>>>(kh, rffWtb, rffb, mask, phiKt, 1);

  dim3 g_gram(M2_ / 64, M2_ / 128, B_ * H_);
  k_gram<<<g_gram, blk, 0, stream>>>(phiKt, G);
  dim3 g_T(1, M2_ / 64, B_ * H_);
  k_tmat<<<g_T, blk, 0, stream>>>(phiKt, vt, Tm);

  k_chol<<<B_ * H_, 256, M2_ * M2_ * sizeof(float), stream>>>(G, Tm, Wsolt);

  dim3 g_out(1, L_ / 64, B_ * H_);
  k_outp<<<g_out, blk, 0, stream>>>(phiQ, Wsolt, yb);

  k_proj<<<g_proj, blk, 0, stream>>>(yb, Wob, mask, bo, nullptr, out, 3);
}
